// ViewTransformer_43679817400611
// MI455X (gfx1250) — compile-verified, matrix operand order checked by reading, not run on hardware
//
#include <hip/hip_runtime.h>
#include <cstdint>

typedef __attribute__((ext_vector_type(16))) _Float16 v16h;
typedef __attribute__((ext_vector_type(8)))  _Float16 v8h;
typedef __attribute__((ext_vector_type(8)))  float    v8f;

constexpr int kH = 128, kW = 128, kHW = kH * kW;
constexpr int kNImg = 12;   // NCAM * B
constexpr int kB = 2;
constexpr int kD = 48;
constexpr float kBnR = 0.9999950000374997f;  // 1/sqrt(1 + 1e-5)

// ---------------------------------------------------------------------------
// Weight packing: (COUT, CIN_SRC, KS, KS) f32 -> [COUT][KS*KS][CIN] f16
// ---------------------------------------------------------------------------
__global__ __launch_bounds__(256) void pack_weights_kernel(
    const float* __restrict__ w, _Float16* __restrict__ wp,
    int cin_src, int cin, int ks2, int total)
{
  int i = blockIdx.x * 256 + threadIdx.x;
  if (i >= total) return;
  int c = i % cin;
  int t = i / cin;
  int kpos = t % ks2;
  int co = t / ks2;
  wp[i] = (_Float16)w[((size_t)co * cin_src + c) * (size_t)ks2 + kpos];
}

__global__ __launch_bounds__(256) void zero_kernel(float* p, int n) {
  int i = blockIdx.x * 256 + threadIdx.x;
  if (i < n) p[i] = 0.f;
}

// ---------------------------------------------------------------------------
// Channels-first -> channels-last transpose: [N][128][HW] -> [N][HW][128]
// Staged through LDS with the gfx1250 async Global->LDS data path
// (GLOBAL_LOAD_ASYNC_TO_LDS_B32, ASYNCcnt), then written back coalesced.
// ---------------------------------------------------------------------------
__global__ __launch_bounds__(256) void transpose_kernel(
    const float* __restrict__ src, float* __restrict__ dst)
{
  __shared__ __align__(16) float tile[32][33];
  const int n  = blockIdx.z;
  const int c0 = blockIdx.y * 32;
  const int p0 = blockIdx.x * 32;
  const int tx = threadIdx.x & 31, ty = threadIdx.x >> 5;
#pragma unroll
  for (int i = 0; i < 32; i += 8) {
    const float* g = src + (size_t)n * 128 * kHW + (size_t)(c0 + ty + i) * kHW + p0 + tx;
    const unsigned ldsa = (unsigned)(uintptr_t)&tile[ty + i][tx];
    asm volatile("global_load_async_to_lds_b32 %0, %1, off"
                 :: "v"(ldsa), "v"(g) : "memory");
  }
  asm volatile("s_wait_asynccnt 0" ::: "memory");
  __syncthreads();
#pragma unroll
  for (int i = 0; i < 32; i += 8)
    dst[(size_t)n * kHW * 128 + (size_t)(p0 + ty + i) * 128 + c0 + tx] = tile[tx][ty + i];
}

// ---------------------------------------------------------------------------
// Fragment load: documented 16-bit 16x32 A layout is two contiguous 16B runs
// per lane ( [8h .. 8h+7] and [16+8h .. 23+8h] ) -> two ds_load_b128.
// ---------------------------------------------------------------------------
__device__ __forceinline__ v16h frag_ld(const _Float16* row, int half) {
  const v8h lo = *(const v8h*)(row + 8 * half);
  const v8h hi = *(const v8h*)(row + 16 + 8 * half);
  v16h r;
#pragma unroll
  for (int i = 0; i < 8; ++i) { r[i] = lo[i]; r[8 + i] = hi[i]; }
  return r;
}

// ---------------------------------------------------------------------------
// Implicit-GEMM conv (KS=3 pad=1 or KS=1 pad=0) + fused bias/BN/ReLU.
// Block: 128 threads (4 waves). Tile: 32 cout x 128 pixels (one image row).
// Each wave: 2 M-tiles x 2 N-tiles -> 4 v_wmma_f32_16x16x32_f16 per k-step.
// ---------------------------------------------------------------------------
template<int CIN, int COUT, int KS, bool PER_IMG_BIAS>
__global__ __launch_bounds__(128) void conv_wmma_kernel(
    const float* __restrict__ src, const _Float16* __restrict__ wp,
    const float* __restrict__ bias, const float* __restrict__ gamma,
    const float* __restrict__ beta, float* __restrict__ dst)
{
  constexpr int PAD  = KS / 2;
  constexpr int CCH  = CIN / 32;
  constexpr int KK   = KS * KS * CCH;
  constexpr int WROW = KS * KS * CIN;

  // stride 40 halves = 80B rows: 16B-aligned, start banks 20*r mod 64 cover
  // all 64 banks -> conflict-free b128 fragment loads.
  __shared__ __align__(16) _Float16 lA[32][40];    // [cout][k]
  __shared__ __align__(16) _Float16 lB[128][40];   // [pixel][k]

  const int tid  = threadIdx.x;
  const int wave = tid >> 5;
  const int lane = tid & 31;
  const int n    = blockIdx.z;
  const int co0  = blockIdx.y * 32;
  const int yrow = blockIdx.x;        // full row of 128 pixels

  const float* srcN = src + (size_t)n * CIN * kHW;

  v8f acc00 = {}, acc01 = {}, acc10 = {}, acc11 = {};

  for (int kt = 0; kt < KK; ++kt) {
    const int c0   = (kt % CCH) * 32;
    const int kpos = kt / CCH;
    const int ky   = kpos / KS;
    const int kx   = kpos % KS;

    // ---- stage A: 32x32 f16 weight tile ----
    const _Float16* wsrc = wp + (size_t)co0 * WROW + (size_t)kpos * CIN + c0;
#pragma unroll
    for (int i = 0; i < 8; ++i) {
      const int idx = tid + 128 * i;
      const int mo = idx >> 5, kc = idx & 31;
      lA[mo][kc] = wsrc[(size_t)mo * WROW + kc];
    }

    // ---- stage B: 128 pixels x 32 channels im2col tile (f32 -> f16) ----
    const int yy = yrow + ky - PAD;
    const bool yok = (yy >= 0) && (yy < kH);
    const float* srow = srcN + (size_t)c0 * kHW + (size_t)yy * kW;
#pragma unroll
    for (int i = 0; i < 32; ++i) {
      const int idx = tid + 128 * i;
      const int p = idx & 127, kc = idx >> 7;
      const int xx = p + kx - PAD;
      float v = 0.f;
      if (yok && xx >= 0 && xx < kW) v = srow[(size_t)kc * kHW + xx];
      lB[p][kc] = (_Float16)v;
    }

    // prefetch next k-slice of the input (global_prefetch)
    if (kt + 1 < KK) {
      const int c0n   = ((kt + 1) % CCH) * 32;
      const int kposn = (kt + 1) / CCH;
      const int kyn   = kposn / KS;
      const int yyn   = yrow + kyn - PAD;
      if (yyn >= 0 && yyn < kH)
        __builtin_prefetch(srcN + (size_t)(c0n + lane) * kHW + (size_t)yyn * kW, 0, 1);
    }
    __syncthreads();

    const int half = lane >> 4;
    const int mrow = lane & 15;
    const int pc   = (wave << 5) + (lane & 15);
    const v16h a0 = frag_ld(&lA[mrow][0], half);
    const v16h a1 = frag_ld(&lA[16 + mrow][0], half);
    const v16h b0 = frag_ld(&lB[pc][0], half);
    const v16h b1 = frag_ld(&lB[pc + 16][0], half);
    acc00 = __builtin_amdgcn_wmma_f32_16x16x32_f16(false, a0, false, b0,
                                                   (short)0, acc00, false, false);
    acc01 = __builtin_amdgcn_wmma_f32_16x16x32_f16(false, a0, false, b1,
                                                   (short)0, acc01, false, false);
    acc10 = __builtin_amdgcn_wmma_f32_16x16x32_f16(false, a1, false, b0,
                                                   (short)0, acc10, false, false);
    acc11 = __builtin_amdgcn_wmma_f32_16x16x32_f16(false, a1, false, b1,
                                                   (short)0, acc11, false, false);
    __syncthreads();
  }

  // ---- epilogue: bias + BN + ReLU; C/D layout: VGPR r -> M = r + 8*half ----
  const int half = lane >> 4;
  const int ncol = lane & 15;
  float* dstN = dst + (size_t)n * COUT * kHW + (size_t)yrow * kW;
#pragma unroll
  for (int mt = 0; mt < 2; ++mt) {
#pragma unroll
    for (int nt = 0; nt < 2; ++nt) {
      const v8f acc = (mt == 0) ? (nt == 0 ? acc00 : acc01)
                                : (nt == 0 ? acc10 : acc11);
      const int p = (wave << 5) + (nt << 4) + ncol;
#pragma unroll
      for (int r = 0; r < 8; ++r) {
        const int m  = r + 8 * half;
        const int co = co0 + 16 * mt + m;
        float v = acc[r];
        const float bi = PER_IMG_BIAS ? bias[n * COUT + co] : bias[co];
        v = (v + bi) * (gamma[co] * kBnR) + beta[co];
        v = fmaxf(v, 0.f);
        dstN[(size_t)co * kHW + p] = v;
      }
    }
  }
}

// ---------------------------------------------------------------------------
// 1x1 conv 32 -> 48 + softmax over the 48 depth bins (per pixel)
// ---------------------------------------------------------------------------
__global__ __launch_bounds__(256) void depth_softmax_kernel(
    const float* __restrict__ h2, const float* __restrict__ w,
    const float* __restrict__ b, float* __restrict__ dp)
{
  __shared__ float sw[kD * 32];
  __shared__ float sb[kD];
  for (int i = threadIdx.x; i < kD * 32; i += 256) sw[i] = w[i];
  if (threadIdx.x < kD) sb[threadIdx.x] = b[threadIdx.x];
  __syncthreads();

  const int n = blockIdx.y;
  const int pix = blockIdx.x * 256 + threadIdx.x;
  const float* inp = h2 + (size_t)n * 32 * kHW + pix;

  float acc[kD];
#pragma unroll
  for (int o = 0; o < kD; ++o) acc[o] = sb[o];
  for (int c = 0; c < 32; ++c) {
    const float x = inp[(size_t)c * kHW];
#pragma unroll
    for (int o = 0; o < kD; ++o) acc[o] = fmaf(sw[o * 32 + c], x, acc[o]);
  }
  float m = acc[0];
#pragma unroll
  for (int o = 1; o < kD; ++o) m = fmaxf(m, acc[o]);
  float s = 0.f;
#pragma unroll
  for (int o = 0; o < kD; ++o) { acc[o] = __expf(acc[o] - m); s += acc[o]; }
  const float inv = 1.f / s;
  float* op = dp + (size_t)n * kD * kHW + pix;
#pragma unroll
  for (int o = 0; o < kD; ++o) op[(size_t)o * kHW] = acc[o] * inv;
}

// ---------------------------------------------------------------------------
// 1x1 conv 32 -> 1 + sigmoid + light mix, reduced to per-image sum (wave32)
// ---------------------------------------------------------------------------
__global__ __launch_bounds__(256) void unc_reduce_kernel(
    const float* __restrict__ u1, const float* __restrict__ w,
    const float* __restrict__ b, const float* __restrict__ light,
    float* __restrict__ sums)
{
  __shared__ float sw[32];
  __shared__ float red[8];
  if (threadIdx.x < 32) sw[threadIdx.x] = w[threadIdx.x];
  __syncthreads();
  const int n = blockIdx.y;
  const int pix = blockIdx.x * 256 + threadIdx.x;
  const float* inp = u1 + (size_t)n * 32 * kHW + pix;
  float acc = b[0];
#pragma unroll
  for (int c = 0; c < 32; ++c) acc = fmaf(sw[c], inp[(size_t)c * kHW], acc);
  float unc = 1.f / (1.f + __expf(-acc));
  const float lf = 1.f - light[n % kB];
  unc = unc + 0.2f * lf * (1.f - unc);
#pragma unroll
  for (int off = 16; off > 0; off >>= 1) unc += __shfl_xor(unc, off, 32);
  if ((threadIdx.x & 31) == 0) red[threadIdx.x >> 5] = unc;
  __syncthreads();
  if (threadIdx.x == 0) {
    float t = 0.f;
    for (int i = 0; i < 8; ++i) t += red[i];
    atomicAdd(&sums[n], t);
  }
}

// ---------------------------------------------------------------------------
// Confidence normalization + fold light channel into the 1x1 lw conv bias
// ---------------------------------------------------------------------------
__global__ __launch_bounds__(256) void finalize_kernel(
    const float* __restrict__ sums, const float* __restrict__ light,
    const float* __restrict__ lwW, const float* __restrict__ lb,
    float* __restrict__ fwgt, float* __restrict__ bias_lw)
{
  __shared__ float conf[kNImg], uncs[kNImg];
  if (threadIdx.x < kNImg) {
    const float us = sums[threadIdx.x] * (1.f / (float)kHW);
    uncs[threadIdx.x] = us;
    conf[threadIdx.x] = 1.f - us;
  }
  __syncthreads();
  if (threadIdx.x < kNImg) {
    const int bb = threadIdx.x % kB;
    float denom = 1e-6f;
    for (int cam = 0; cam < 6; ++cam) denom += conf[cam * kB + bb];
    // bev scale (1-unc_s) * normalized conf  (valid_mean == 1)
    fwgt[threadIdx.x] = (1.f - uncs[threadIdx.x]) * conf[threadIdx.x] / denom;
  }
  for (int i = threadIdx.x; i < kB * 256; i += 256) {
    const int bb = i >> 8, co = i & 255;
    bias_lw[i] = lb[co] + lwW[co * 257 + 256] * light[bb];
  }
}

// ---------------------------------------------------------------------------
// BEV lift: per BEV pixel, sum over cams/depth bins of dp-weighted bilinear
// samples of the channels-last camera features. One wave covers all 128
// channels (float4/lane -> perfectly coalesced 512B corner loads).
// ---------------------------------------------------------------------------
__global__ __launch_bounds__(256) void bev_sample_kernel(
    const float* __restrict__ featsT, const float* __restrict__ dp,
    const float* __restrict__ fwgt, float* __restrict__ img_bev)
{
  const int lane = threadIdx.x & 31;
  const int wave = threadIdx.x >> 5;
  const int b = blockIdx.y;
  const int pix0 = blockIdx.x * 32 + wave * 4;

  float4 acc[4];
#pragma unroll
  for (int i = 0; i < 4; ++i) acc[i] = make_float4(0.f, 0.f, 0.f, 0.f);

  for (int cam = 0; cam < 6; ++cam) {
    const int n = cam * kB + b;
    const float fw = fwgt[n];
    const float* ft = featsT + (size_t)n * kHW * 128 + lane * 4;
    const float* dpn = dp + (size_t)n * kD * kHW;
#pragma unroll
    for (int pi = 0; pi < 4; ++pi) {
      const int pix = pix0 + pi;
      const int nx = pix & 127;
      const int ny = pix >> 7;
      const float xc = -50.8f + 0.8f * (float)nx;
      const float yc = -50.8f + 0.8f * (float)ny;
#pragma unroll 1
      for (int d = 0; d < kD; ++d) {
        const float invz = 1.f / (1.f + (float)d + 1e-6f);
        const float sc = invz * (127.f / 512.f);
        const float xi = xc * sc;
        const float yi = yc * sc;
        const float xf = floorf(xi), yf = floorf(yi);
        const int ix = (int)xf, iy = (int)yf;
        if (ix < -1 || ix > 127 || iy < -1 || iy > 127) continue;
        const float wx1 = xi - xf, wy1 = yi - yf;
        const float pw = fw * dpn[(size_t)d * kHW + pix];
        const float w00 = pw * (1.f - wy1) * (1.f - wx1);
        const float w01 = pw * (1.f - wy1) * wx1;
        const float w10 = pw * wy1 * (1.f - wx1);
        const float w11 = pw * wy1 * wx1;
        const bool vx0 = (ix >= 0) && (ix <= 127);
        const bool vx1 = (ix + 1 <= 127);
        const bool vy0 = (iy >= 0) && (iy <= 127);
        const bool vy1 = (iy + 1 <= 127);
        auto corner = [&](int yyc, int xxc, float wgt) {
          const float4 s = *(const float4*)(ft + (size_t)(yyc * 128 + xxc) * 128);
          acc[pi].x = fmaf(wgt, s.x, acc[pi].x);
          acc[pi].y = fmaf(wgt, s.y, acc[pi].y);
          acc[pi].z = fmaf(wgt, s.z, acc[pi].z);
          acc[pi].w = fmaf(wgt, s.w, acc[pi].w);
        };
        if (vy0 && vx0) corner(iy, ix, w00);
        if (vy0 && vx1) corner(iy, ix + 1, w01);
        if (vy1 && vx0) corner(iy + 1, ix, w10);
        if (vy1 && vx1) corner(iy + 1, ix + 1, w11);
      }
    }
  }
#pragma unroll
  for (int pi = 0; pi < 4; ++pi) {
    const int pix = pix0 + pi;
    float* o = img_bev + (size_t)b * 128 * kHW + pix;
    o[(size_t)(4 * lane + 0) * kHW] = acc[pi].x;
    o[(size_t)(4 * lane + 1) * kHW] = acc[pi].y;
    o[(size_t)(4 * lane + 2) * kHW] = acc[pi].z;
    o[(size_t)(4 * lane + 3) * kHW] = acc[pi].w;
  }
}

// ---------------------------------------------------------------------------
extern "C" void kernel_launch(void* const* d_in, const int* in_sizes, int n_in,
                              void* d_out, int out_size, void* d_ws, size_t ws_size,
                              hipStream_t stream)
{
  (void)in_sizes; (void)n_in; (void)out_size; (void)ws_size;

  const float* img_feats = (const float*)d_in[0];
  const float* light = (const float*)d_in[1];
  const float* dw1 = (const float*)d_in[2];
  const float* db1 = (const float*)d_in[3];
  const float* dg1 = (const float*)d_in[4];
  const float* dbe1 = (const float*)d_in[5];
  const float* dw2 = (const float*)d_in[6];
  const float* db2 = (const float*)d_in[7];
  const float* dg2 = (const float*)d_in[8];
  const float* dbe2 = (const float*)d_in[9];
  const float* dw3 = (const float*)d_in[10];
  const float* db3 = (const float*)d_in[11];
  const float* uw1 = (const float*)d_in[12];
  const float* ub1 = (const float*)d_in[13];
  const float* ug1 = (const float*)d_in[14];
  const float* ube1 = (const float*)d_in[15];
  const float* uw2 = (const float*)d_in[16];
  const float* ub2 = (const float*)d_in[17];
  const float* ew1 = (const float*)d_in[18];
  const float* eb1 = (const float*)d_in[19];
  const float* eg1 = (const float*)d_in[20];
  const float* ebe1 = (const float*)d_in[21];
  const float* ew2 = (const float*)d_in[22];
  const float* eb2 = (const float*)d_in[23];
  const float* eg2 = (const float*)d_in[24];
  const float* ebe2 = (const float*)d_in[25];
  const float* lwW = (const float*)d_in[26];
  const float* lb = (const float*)d_in[27];
  const float* lg = (const float*)d_in[28];
  const float* lbe = (const float*)d_in[29];

  char* ws = (char*)d_ws;
  size_t off = 0;
  auto alloc = [&](size_t bytes) -> void* {
    void* p = ws + off;
    off += (bytes + 255) & ~(size_t)255;
    return p;
  };

  float* feats_t = (float*)alloc((size_t)kNImg * kHW * 128 * 4);
  float* h1      = (float*)alloc((size_t)kNImg * 64 * kHW * 4);
  float* h2      = (float*)alloc((size_t)kNImg * 32 * kHW * 4);
  float* dpbuf   = (float*)alloc((size_t)kNImg * kD * kHW * 4);
  float* img_bev = (float*)alloc((size_t)kB * 128 * kHW * 4);
  float* e1      = (float*)alloc((size_t)kB * 256 * kHW * 4);
  float* e2      = (float*)alloc((size_t)kB * 256 * kHW * 4);
  float* sums    = (float*)alloc(kNImg * 4);
  float* fwgt    = (float*)alloc(kNImg * 4);
  float* bias_lw = (float*)alloc((size_t)kB * 256 * 4);
  _Float16* wp_dw1 = (_Float16*)alloc((size_t)64 * 9 * 128 * 2);
  _Float16* wp_dw2 = (_Float16*)alloc((size_t)32 * 9 * 64 * 2);
  _Float16* wp_uw1 = (_Float16*)alloc((size_t)32 * 9 * 128 * 2);
  _Float16* wp_ew1 = (_Float16*)alloc((size_t)256 * 9 * 128 * 2);
  _Float16* wp_ew2 = (_Float16*)alloc((size_t)256 * 9 * 256 * 2);
  _Float16* wp_lw  = (_Float16*)alloc((size_t)256 * 256 * 2);
  float* u1 = h1;  // h1 is dead after dw2 -> reuse for the uncertainty branch

  auto packLaunch = [&](const float* w, _Float16* wp, int cin_src, int cin,
                        int ks, int cout) {
    int total = cout * ks * ks * cin;
    pack_weights_kernel<<<(total + 255) / 256, 256, 0, stream>>>(
        w, wp, cin_src, cin, ks * ks, total);
  };
  packLaunch(dw1, wp_dw1, 128, 128, 3, 64);
  packLaunch(dw2, wp_dw2, 64, 64, 3, 32);
  packLaunch(uw1, wp_uw1, 128, 128, 3, 32);
  packLaunch(ew1, wp_ew1, 128, 128, 3, 256);
  packLaunch(ew2, wp_ew2, 256, 256, 3, 256);
  packLaunch(lwW, wp_lw, 257, 256, 1, 256);

  zero_kernel<<<1, 256, 0, stream>>>(sums, kNImg);

  transpose_kernel<<<dim3(kHW / 32, 4, kNImg), 256, 0, stream>>>(img_feats, feats_t);

  // depth branch
  conv_wmma_kernel<128, 64, 3, false><<<dim3(kH, 2, kNImg), 128, 0, stream>>>(
      img_feats, wp_dw1, db1, dg1, dbe1, h1);
  conv_wmma_kernel<64, 32, 3, false><<<dim3(kH, 1, kNImg), 128, 0, stream>>>(
      h1, wp_dw2, db2, dg2, dbe2, h2);
  depth_softmax_kernel<<<dim3(kHW / 256, kNImg), 256, 0, stream>>>(h2, dw3, db3, dpbuf);

  // uncertainty branch
  conv_wmma_kernel<128, 32, 3, false><<<dim3(kH, 1, kNImg), 128, 0, stream>>>(
      img_feats, wp_uw1, ub1, ug1, ube1, u1);
  unc_reduce_kernel<<<dim3(kHW / 256, kNImg), 256, 0, stream>>>(u1, uw2, ub2, light, sums);
  finalize_kernel<<<1, 256, 0, stream>>>(sums, light, lwW, lb, fwgt, bias_lw);

  // BEV lift + fuse
  bev_sample_kernel<<<dim3(kHW / 32, kB), 256, 0, stream>>>(feats_t, dpbuf, fwgt, img_bev);

  // encoder
  conv_wmma_kernel<128, 256, 3, false><<<dim3(kH, 8, kB), 128, 0, stream>>>(
      img_bev, wp_ew1, eb1, eg1, ebe1, e1);
  conv_wmma_kernel<256, 256, 3, false><<<dim3(kH, 8, kB), 128, 0, stream>>>(
      e1, wp_ew2, eb2, eg2, ebe2, e2);
  conv_wmma_kernel<256, 256, 1, true><<<dim3(kH, 8, kB), 128, 0, stream>>>(
      e2, wp_lw, bias_lw, lg, lbe, (float*)d_out);
}
